// LatentDecoder_68135361184483
// MI455X (gfx1250) — matrix operand order repaired and checked by reading, not compiled, not run on hardware
//
#include <hip/hip_runtime.h>
#include <hip/hip_bf16.h>

typedef __bf16 bf16;
typedef __attribute__((ext_vector_type(16))) __bf16 v16bf;
typedef __attribute__((ext_vector_type(8)))  __bf16 v8bf;
typedef __attribute__((ext_vector_type(4)))  __bf16 v4bf;
typedef __attribute__((ext_vector_type(8)))  float  v8f;

// ---------------- WMMA fragment loaders (CDNA5 16x16x32 bf16) ----------------
// A fragment (16x32, M x K): lane&15 = row; e[0..7]=K[kb..kb+7], e[8..15]=K[kb+16..kb+23], kb=(lane>>4)*8
// row pointer must be 16B aligned; both chunks are contiguous 16B -> ds_load_b128 x2
__device__ inline v16bf ld_frag_a_bf16(const bf16* __restrict__ row, int lane) {
  const int kb = (lane >> 4) * 8;
  const v8bf lo = *(const v8bf*)(row + kb);
  const v8bf hi = *(const v8bf*)(row + kb + 16);
  return __builtin_shufflevector(lo, hi, 0, 1, 2, 3, 4, 5, 6, 7,
                                 8, 9, 10, 11, 12, 13, 14, 15);
}
// A fragment sourced from f32 LDS (attention probabilities), converted on the fly
__device__ inline v16bf ld_frag_a_f32(const float* __restrict__ row, int lane) {
  const int kb = (lane >> 4) * 8;
  const float4 a0 = *(const float4*)(row + kb);
  const float4 a1 = *(const float4*)(row + kb + 4);
  const float4 b0 = *(const float4*)(row + kb + 16);
  const float4 b1 = *(const float4*)(row + kb + 20);
  v16bf f;
  f[0] = (bf16)a0.x; f[1] = (bf16)a0.y; f[2]  = (bf16)a0.z; f[3]  = (bf16)a0.w;
  f[4] = (bf16)a1.x; f[5] = (bf16)a1.y; f[6]  = (bf16)a1.z; f[7]  = (bf16)a1.w;
  f[8] = (bf16)b0.x; f[9] = (bf16)b0.y; f[10] = (bf16)b0.z; f[11] = (bf16)b0.w;
  f[12] = (bf16)b1.x; f[13] = (bf16)b1.y; f[14] = (bf16)b1.z; f[15] = (bf16)b1.w;
  return f;
}
// B fragment (32x16, K x N) stored as N-rows along K (W row-major):
// lane&15 = column; e[0..7]=K[kb..kb+7], e[8..15]=K[kb+16..kb+23], kb=(lane>>4)*8 -> ds_load_b128 x2
__device__ inline v16bf ld_frag_b_bf16(const bf16* __restrict__ row, int lane) {
  const int kb = (lane >> 4) * 8;
  const v8bf lo = *(const v8bf*)(row + kb);
  const v8bf hi = *(const v8bf*)(row + kb + 16);
  return __builtin_shufflevector(lo, hi, 0, 1, 2, 3, 4, 5, 6, 7,
                                 8, 9, 10, 11, 12, 13, 14, 15);
}
__device__ inline v4bf cvt4(const float4 v) {
  v4bf r; r[0] = (bf16)v.x; r[1] = (bf16)v.y; r[2] = (bf16)v.z; r[3] = (bf16)v.w;
  return r;
}

// ---------------- Generic tiled GEMM: C[M,N] = A[M,K] @ W[N,K]^T + bias ----------------
#define BM 128
#define BN 64
#define BK 32
#define LDT (BK + 8)   // 40 bf16 = 80B row stride, multiple of 16B

__global__ void __launch_bounds__(256)
k_gemm_bf16(const float* __restrict__ A, const float* __restrict__ W,
            const float* __restrict__ bias, float* __restrict__ C,
            int M, int N, int K, int relu) {
  __shared__ __align__(16) bf16 As[BM][LDT];
  __shared__ __align__(16) bf16 Ws[BN][LDT];
  const int tid = threadIdx.x, lane = tid & 31, wv = tid >> 5;
  const int m0 = blockIdx.y * BM, n0 = blockIdx.x * BN;

  v8f acc[4];
#pragma unroll
  for (int t = 0; t < 4; ++t)
#pragma unroll
    for (int r = 0; r < 8; ++r) acc[t][r] = 0.f;

  for (int k0 = 0; k0 < K; k0 += BK) {
    // prefetch next K tile (one 128B cacheline covers a full BK row of f32)
    if (k0 + BK < K) {
      if (tid < BM)
        __builtin_prefetch(A + (size_t)(m0 + tid) * K + k0 + BK, 0, 3);
      else if (tid - BM < BN && n0 + (tid - BM) < N)
        __builtin_prefetch(W + (size_t)(n0 + tid - BM) * K + k0 + BK, 0, 3);
    }
    // stage A (f32 -> bf16, 8B LDS stores)
    for (int idx = tid; idx < BM * (BK / 4); idx += 256) {
      const int r = idx >> 3, c = (idx & 7) * 4;
      const float4 v = *(const float4*)(A + (size_t)(m0 + r) * K + k0 + c);
      *(v4bf*)&As[r][c] = cvt4(v);
    }
    for (int idx = tid; idx < BN * (BK / 4); idx += 256) {
      const int r = idx >> 3, c = (idx & 7) * 4;
      v4bf pk;
      if (n0 + r < N) {
        pk = cvt4(*(const float4*)(W + (size_t)(n0 + r) * K + k0 + c));
      } else {
        pk[0] = (bf16)0.f; pk[1] = (bf16)0.f; pk[2] = (bf16)0.f; pk[3] = (bf16)0.f;
      }
      *(v4bf*)&Ws[r][c] = pk;
    }
    __syncthreads();
    // preload all fragments, then fire WMMAs back-to-back (single dscnt wait)
    const v16bf a = ld_frag_a_bf16(&As[wv * 16 + (lane & 15)][0], lane);
    v16bf b[4];
#pragma unroll
    for (int t = 0; t < 4; ++t)
      b[t] = ld_frag_b_bf16(&Ws[t * 16 + (lane & 15)][0], lane);
#pragma unroll
    for (int t = 0; t < 4; ++t)
      acc[t] = __builtin_amdgcn_wmma_f32_16x16x32_bf16(false, a, false, b[t],
                                                       (short)0, acc[t], false, false);
    __syncthreads();
  }

  const int rbase = m0 + wv * 16 + ((lane >> 4) << 3);
  const int cw = lane & 15;
#pragma unroll
  for (int t = 0; t < 4; ++t) {
    const int col = n0 + t * 16 + cw;
    if (col < N) {
      const float bv = bias ? bias[col] : 0.f;
#pragma unroll
      for (int r = 0; r < 8; ++r) {
        float v = acc[t][r] + bv;
        if (relu) v = fmaxf(v, 0.f);
        C[(size_t)(rbase + r) * N + col] = v;
      }
    }
  }
}

// ---------------- Self-attention: one block per (head, batch) ----------------
// qkv: [B,128,1536] (Q|K|V each 512). out: [B,128,512]
__global__ void __launch_bounds__(256)
k_attention(const float* __restrict__ qkv, float* __restrict__ out) {
  extern __shared__ __align__(16) char smem[];
  bf16*  Qs = (bf16*)smem;                 // 128x64  (16KB)
  bf16*  Ks = (bf16*)(smem + 16384);       // 128x64  (16KB)
  bf16*  Vt = (bf16*)(smem + 32768);       // 64x128  (16KB), Vt[d][k]=V[k][d]
  float* S  = (float*)(smem + 49152);      // 128x132 (67.6KB)
  const int h = blockIdx.x, b = blockIdx.y;
  const int tid = threadIdx.x, lane = tid & 31, wv = tid >> 5;
  const size_t base = (size_t)b * 128 * 1536 + h * 64;

  // stage Q,K (vectorized) and V transposed (scalar, strided by nature)
  for (int idx = tid; idx < 128 * 16; idx += 256) {
    const int l = idx >> 4, d = (idx & 15) * 4;
    const float* p = qkv + base + (size_t)l * 1536 + d;
    const float4 q = *(const float4*)(p);
    const float4 k = *(const float4*)(p + 512);
    const float4 v = *(const float4*)(p + 1024);
    *(v4bf*)&Qs[l * 64 + d] = cvt4(q);
    *(v4bf*)&Ks[l * 64 + d] = cvt4(k);
    Vt[(d + 0) * 128 + l] = (bf16)v.x;
    Vt[(d + 1) * 128 + l] = (bf16)v.y;
    Vt[(d + 2) * 128 + l] = (bf16)v.z;
    Vt[(d + 3) * 128 + l] = (bf16)v.w;
  }
  __syncthreads();

  { // S = (Q @ K^T) * rsqrt(hd) + causal mask
    v8f acc[8];
#pragma unroll
    for (int t = 0; t < 8; ++t)
#pragma unroll
      for (int r = 0; r < 8; ++r) acc[t][r] = 0.f;
#pragma unroll
    for (int kt = 0; kt < 2; ++kt) {
      const v16bf a = ld_frag_a_bf16(&Qs[(wv * 16 + (lane & 15)) * 64 + kt * 32], lane);
      v16bf bb[8];
#pragma unroll
      for (int t = 0; t < 8; ++t)
        bb[t] = ld_frag_b_bf16(&Ks[(t * 16 + (lane & 15)) * 64 + kt * 32], lane);
#pragma unroll
      for (int t = 0; t < 8; ++t)
        acc[t] = __builtin_amdgcn_wmma_f32_16x16x32_bf16(false, a, false, bb[t],
                                                         (short)0, acc[t], false, false);
    }
    const int rbase = wv * 16 + ((lane >> 4) << 3);
#pragma unroll
    for (int t = 0; t < 8; ++t) {
      const int col = t * 16 + (lane & 15);
#pragma unroll
      for (int r = 0; r < 8; ++r) {
        float v = acc[t][r] * 0.125f;           // 1/sqrt(64)
        if (col > rbase + r) v += -1e9f;        // causal additive mask
        S[(rbase + r) * 132 + col] = v;
      }
    }
  }
  __syncthreads();

  if (tid < 128) { // row softmax
    float* row = &S[tid * 132];
    float m = -3.4e38f;
    for (int j = 0; j < 128; ++j) m = fmaxf(m, row[j]);
    float s = 0.f;
    for (int j = 0; j < 128; ++j) { const float e = __expf(row[j] - m); row[j] = e; s += e; }
    const float inv = 1.f / s;
    for (int j = 0; j < 128; ++j) row[j] *= inv;
  }
  __syncthreads();

  { // O = P @ V  (M=128, N=64, K=128); A from f32 LDS converted on the fly
    v8f acc[4];
#pragma unroll
    for (int t = 0; t < 4; ++t)
#pragma unroll
      for (int r = 0; r < 8; ++r) acc[t][r] = 0.f;
#pragma unroll
    for (int kt = 0; kt < 4; ++kt) {
      const v16bf a = ld_frag_a_f32(&S[(wv * 16 + (lane & 15)) * 132 + kt * 32], lane);
      v16bf bb[4];
#pragma unroll
      for (int t = 0; t < 4; ++t)
        bb[t] = ld_frag_b_bf16(&Vt[(t * 16 + (lane & 15)) * 128 + kt * 32], lane);
#pragma unroll
      for (int t = 0; t < 4; ++t)
        acc[t] = __builtin_amdgcn_wmma_f32_16x16x32_bf16(false, a, false, bb[t],
                                                         (short)0, acc[t], false, false);
    }
    const int rbase = wv * 16 + ((lane >> 4) << 3);
#pragma unroll
    for (int t = 0; t < 4; ++t) {
      const int col = h * 64 + t * 16 + (lane & 15);
#pragma unroll
      for (int r = 0; r < 8; ++r)
        out[((size_t)b * 128 + rbase + r) * 512 + col] = acc[t][r];
    }
  }
}

// ---------------- Residual + LayerNorm over D=512 (one block per token) ----------------
__global__ void __launch_bounds__(256)
k_res_ln(float* __restrict__ x, const float* __restrict__ y,
         const float* __restrict__ g, const float* __restrict__ be, int bcast) {
  __shared__ float red[256], red2[256], stats[2];
  const int row = blockIdx.x, tid = threadIdx.x;
  float* xr = x + (size_t)row * 512;
  const float* yr = bcast ? (y + (size_t)(row >> 7) * 512) : (y + (size_t)row * 512);
  const float v0 = xr[tid] + yr[tid];
  const float v1 = xr[tid + 256] + yr[tid + 256];
  red[tid] = v0 + v1; red2[tid] = v0 * v0 + v1 * v1;
  __syncthreads();
  for (int off = 128; off > 0; off >>= 1) {
    if (tid < off) { red[tid] += red[tid + off]; red2[tid] += red2[tid + off]; }
    __syncthreads();
  }
  if (tid == 0) {
    const float mu = red[0] * (1.f / 512.f);
    const float var = red2[0] * (1.f / 512.f) - mu * mu;
    stats[0] = mu; stats[1] = rsqrtf(var + 1e-5f);
  }
  __syncthreads();
  const float mu = stats[0], rs = stats[1];
  xr[tid]       = (v0 - mu) * rs * g[tid] + be[tid];
  xr[tid + 256] = (v1 - mu) * rs * g[tid + 256] + be[tid + 256];
}

// ---------------- mem = LN(GELU(z @ lp_w^T + lp_b)) ----------------
__global__ void __launch_bounds__(512)
k_latent(const float* __restrict__ z, const float* __restrict__ w,
         const float* __restrict__ bia, const float* __restrict__ g,
         const float* __restrict__ be, float* __restrict__ mem) {
  __shared__ float zs[256];
  __shared__ float red[512], red2[512], stats[2];
  const int b = blockIdx.x, d = threadIdx.x;
  if (d < 256) zs[d] = z[b * 256 + d];
  __syncthreads();
  float acc = bia[d];
  const float* wr = w + (size_t)d * 256;
  for (int k = 0; k < 256; ++k) acc += zs[k] * wr[k];
  const float hg = 0.5f * acc * (1.f + erff(acc * 0.7071067811865475f)); // exact GELU
  red[d] = hg; red2[d] = hg * hg;
  __syncthreads();
  for (int off = 256; off > 0; off >>= 1) {
    if (d < off) { red[d] += red[d + off]; red2[d] += red2[d + off]; }
    __syncthreads();
  }
  if (d == 0) {
    const float mu = red[0] * (1.f / 512.f);
    const float var = red2[0] * (1.f / 512.f) - mu * mu;
    stats[0] = mu; stats[1] = rsqrtf(var + 1e-5f);
  }
  __syncthreads();
  mem[(size_t)b * 512 + d] = (hg - stats[0]) * stats[1] * g[d] + be[d];
}

// ---------------- cross-attn vector: Lk==1 => softmax==1 => o = V ----------------
// ca[l][b] = (mem[b] @ Wv_l^T + bv_l) @ Wo_l^T + bo_l
__global__ void __launch_bounds__(512)
k_ca_vec(const float* __restrict__ mem, const float* __restrict__ qkv_w,
         const float* __restrict__ qkv_b, const float* __restrict__ out_w,
         const float* __restrict__ out_b, float* __restrict__ ca) {
  __shared__ float ms[512], vs[512];
  const int b = blockIdx.x, l = blockIdx.y, d = threadIdx.x;
  ms[d] = mem[(size_t)b * 512 + d];
  __syncthreads();
  const float* wv = qkv_w + (size_t)l * 1536 * 512 + (size_t)(1024 + d) * 512;
  float acc = qkv_b[l * 1536 + 1024 + d];
  for (int k = 0; k < 512; ++k) acc += ms[k] * wv[k];
  vs[d] = acc;
  __syncthreads();
  const float* wo = out_w + (size_t)l * 512 * 512 + (size_t)d * 512;
  float o = out_b[l * 512 + d];
  for (int k = 0; k < 512; ++k) o += vs[k] * wo[k];
  ca[((size_t)l * 32 + b) * 512 + d] = o;
}

// ---------------- embedding ----------------
__global__ void __launch_bounds__(256)
k_embed(const int* __restrict__ ids, const float* __restrict__ tok,
        const float* __restrict__ pos, float* __restrict__ x) {
  const int row = blockIdx.x, tid = threadIdx.x, id = ids[row], l = row & 127;
#pragma unroll
  for (int i = 0; i < 2; ++i) {
    const int d = tid + i * 256;
    x[(size_t)row * 512 + d] = tok[(size_t)id * 512 + d] + pos[(size_t)l * 512 + d];
  }
}

// ---------------- orchestration ----------------
extern "C" void kernel_launch(void* const* d_in, const int* in_sizes, int n_in,
                              void* d_out, int out_size, void* d_ws, size_t ws_size,
                              hipStream_t stream) {
  const float* z       = (const float*)d_in[0];
  const int*   ids     = (const int*)d_in[1];
  const float* tok_emb = (const float*)d_in[2];
  const float* pos_emb = (const float*)d_in[3];
  const float* lp_w    = (const float*)d_in[4];
  const float* lp_b    = (const float*)d_in[5];
  const float* lp_g    = (const float*)d_in[6];
  const float* lp_be   = (const float*)d_in[7];
  const float* sa_qkv_w = (const float*)d_in[8];
  const float* sa_qkv_b = (const float*)d_in[9];
  const float* sa_out_w = (const float*)d_in[10];
  const float* sa_out_b = (const float*)d_in[11];
  const float* ca_qkv_w = (const float*)d_in[12];
  const float* ca_qkv_b = (const float*)d_in[13];
  const float* ca_out_w = (const float*)d_in[14];
  const float* ca_out_b = (const float*)d_in[15];
  const float* ln1_g = (const float*)d_in[16];
  const float* ln1_b = (const float*)d_in[17];
  const float* ln2_g = (const float*)d_in[18];
  const float* ln2_b = (const float*)d_in[19];
  const float* ln3_g = (const float*)d_in[20];
  const float* ln3_b = (const float*)d_in[21];
  const float* ff1_w = (const float*)d_in[22];
  const float* ff1_b = (const float*)d_in[23];
  const float* ff2_w = (const float*)d_in[24];
  const float* ff2_b = (const float*)d_in[25];
  const float* op_w  = (const float*)d_in[26];
  const float* op_b  = (const float*)d_in[27];

  const int M = 32 * 128, D = 512, V = 50257, NL = 4, DF = 2048;

  char* ws = (char*)d_ws;
  float* mem  = (float*)(ws + 0);           //  64 KB
  float* ca   = (float*)(ws + 65536);       // 256 KB
  float* x    = (float*)(ws + 327680);      //   8 MB
  float* big  = (float*)(ws + 8716288);     //  33.5 MB (qkv / ff1, disjoint lifetimes)
  float* attn = (float*)(ws + 42270720);    //   8 MB
  float* tmp  = (float*)(ws + 50659328);    //   8 MB

  k_latent<<<32, 512, 0, stream>>>(z, lp_w, lp_b, lp_g, lp_be, mem);
  k_ca_vec<<<dim3(32, NL), 512, 0, stream>>>(mem, ca_qkv_w, ca_qkv_b, ca_out_w, ca_out_b, ca);
  k_embed<<<M, 256, 0, stream>>>(ids, tok_emb, pos_emb, x);

  const size_t attn_smem = 16384 * 3 + 128 * 132 * 4;
  for (int l = 0; l < NL; ++l) {
    // self-attention
    k_gemm_bf16<<<dim3(3 * D / BN, M / BM), 256, 0, stream>>>(
        x, sa_qkv_w + (size_t)l * 3 * D * D, sa_qkv_b + (size_t)l * 3 * D, big, M, 3 * D, D, 0);
    k_attention<<<dim3(8, 32), 256, attn_smem, stream>>>(big, attn);
    k_gemm_bf16<<<dim3(D / BN, M / BM), 256, 0, stream>>>(
        attn, sa_out_w + (size_t)l * D * D, sa_out_b + (size_t)l * D, tmp, M, D, D, 0);
    k_res_ln<<<M, 256, 0, stream>>>(x, tmp, ln1_g + l * D, ln1_b + l * D, 0);
    // cross-attention (analytically collapsed; broadcast per batch)
    k_res_ln<<<M, 256, 0, stream>>>(x, ca + (size_t)l * 32 * D, ln2_g + l * D, ln2_b + l * D, 1);
    // FFN
    k_gemm_bf16<<<dim3(DF / BN, M / BM), 256, 0, stream>>>(
        x, ff1_w + (size_t)l * DF * D, ff1_b + (size_t)l * DF, big, M, DF, D, 1);
    k_gemm_bf16<<<dim3(D / BN, M / BM), 256, 0, stream>>>(
        big, ff2_w + (size_t)l * D * DF, ff2_b + (size_t)l * D, tmp, M, D, DF, 0);
    k_res_ln<<<M, 256, 0, stream>>>(x, tmp, ln3_g + l * D, ln3_b + l * D, 0);
  }

  // vocab projection (dominant GEMM; op_w fits in 192MB L2 across M-tiles)
  k_gemm_bf16<<<dim3((V + BN - 1) / BN, M / BM), 256, 0, stream>>>(
      x, op_w, op_b, (float*)d_out, M, V, D, 0);
}
